// GraphAttention_73753178407510
// MI455X (gfx1250) — matrix operand order repaired and checked by reading, not compiled, not run on hardware
//
#include <hip/hip_runtime.h>
#include <stdint.h>

// ---------------------------------------------------------------------------
// Types for CDNA5 WMMA / TDM
// ---------------------------------------------------------------------------
typedef __attribute__((ext_vector_type(16))) __bf16 v16bf;
typedef __attribute__((ext_vector_type(8)))  float  v8f;
typedef __attribute__((ext_vector_type(4)))  unsigned int u32x4;
typedef __attribute__((ext_vector_type(8)))  int i32x8;
typedef __attribute__((ext_vector_type(4)))  int i32x4;

#define TILE_M 128
#define TILE_N 128
#define TILE_K 32
#define LDK    40   // padded LDS row stride (elements): bank-conflict free

#if defined(__gfx1250__) && __has_builtin(__builtin_amdgcn_tensor_load_to_lds)
  #define USE_TDM 1
  #if __has_include(<hip/amd_detail/amd_gfx1250_TDM.h>)
    #define TDM_SIX_ARG 1
  #else
    #define TDM_SIX_ARG 0
  #endif
#else
  #define USE_TDM 0
#endif

union FragU {
    uint4 u[2];
    v16bf v;
};

// A fragment (16x32 M x K, bf16): lane L holds row m = L&15.
// VGPR0..3 = K = h*8 .. h*8+7, VGPR4..7 = K = 16+h*8 .. +7  (h = L>>4)
__device__ inline v16bf load_a_frag(const __bf16* smem, int lane) {
    int m = lane & 15;
    int h = lane >> 4;
    const __bf16* row = smem + m * LDK + h * 8;
    FragU f;
    f.u[0] = *(const uint4*)(row);
    f.u[1] = *(const uint4*)(row + 16);
    return f.v;
}

// B fragment (32x16 K x N, bf16): lane L holds column n = L&15,
// VGPR0..3 = K = h*8 .. h*8+7, VGPR4..7 = K = 16+h*8 .. +7. smem holds W tile as [n][k] row-major.
__device__ inline v16bf load_b_frag(const __bf16* smem, int lane) {
    int n = lane & 15;
    int h = lane >> 4;
    const __bf16* p = smem + n * LDK + h * 8;
    FragU f;
    f.u[0] = *(const uint4*)(p);
    f.u[1] = *(const uint4*)(p + 16);
    return f.v;
}

#if USE_TDM
// Issue a TDM DMA: load a (rows x TILE_K) bf16 tile (row stride = row_len
// elements in memory) into LDS at lds_off, inserting 16B of padding after
// every 64B row so the LDS layout has a 40-element (80B) row stride.
// D# built per CDNA5 ISA section 8 (groups 0/1; 2-D tensor -> groups 2/3 = 0).
__device__ inline void tdm_load_tile(unsigned lds_off, const __bf16* gptr,
                                     int rows, int row_len, int nrows_tensor) {
    unsigned long long ga = (unsigned long long)(uintptr_t)gptr;
    u32x4 g0;
    g0.x = 1u;                                       // count=1 valid descriptor
    g0.y = lds_off;                                  // lds_addr (bytes)
    g0.z = (unsigned)(ga & 0xFFFFFFFFu);             // global_addr[31:0]
    g0.w = (unsigned)((ga >> 32) & 0x01FFFFFFu)      // global_addr[56:32]
         | (2u << 30);                               // type = 2 ("image")
    i32x8 g1;
    g1[0] = (int)((1u << 16)                         // data_size = 2 bytes
                | (1u << 20)                         // pad_enable
                | (3u << 22)                         // pad_interval: 16 DWORDs
                | (3u << 25));                       // pad_amount:   4 DWORDs
    g1[1] = (int)((unsigned)(row_len & 0xFFFF) << 16);          // tensor_dim0 lo
    g1[2] = (int)(((unsigned)(row_len >> 16) & 0xFFFFu)         // tensor_dim0 hi
                | ((unsigned)(nrows_tensor & 0xFFFF) << 16));   // tensor_dim1 lo
    g1[3] = (int)(((unsigned)(nrows_tensor >> 16) & 0xFFFFu)    // tensor_dim1 hi
                | ((unsigned)TILE_K << 16));                    // tile_dim0 = 32
    g1[4] = rows;                                    // tile_dim1 (tile_dim2 = 0)
    g1[5] = row_len;                                 // tensor_dim0_stride lo
    g1[6] = 0;                                       // stride hi / dim1_stride lo
    g1[7] = 0;
    i32x4 z4 = {0, 0, 0, 0};
#if TDM_SIX_ARG
    i32x8 z8 = {0, 0, 0, 0, 0, 0, 0, 0};
    __builtin_amdgcn_tensor_load_to_lds(g0, g1, z4, z4, z8, 0);
#else
    __builtin_amdgcn_tensor_load_to_lds(g0, g1, z4, z4, 0);
#endif
}

__device__ inline unsigned lds_byte_off(const void* p) {
    // LDS aperture: flat addr low 32 bits == LDS byte offset (ISA 10.2)
    return (unsigned)(uintptr_t)p;
}
#endif // USE_TDM

// ---------------------------------------------------------------------------
// Y[m][n] = sum_k A[m][k] * W[n][k] + bias[n]     (A: MxK bf16, W: NxK bf16)
// Block: 256 threads (8 waves). Block tile 128x128, K step 32.
// Wave (wm,wn): wm 0..3 -> 32 rows, wn 0..1 -> 64 cols => 2x4 WMMA tiles.
// TDM path: double-buffered LDS, tiles DMA'd by tensor_load_to_lds, one
// barrier per K step; wave 0 owns TENSORcnt.
// ---------------------------------------------------------------------------
__global__ __launch_bounds__(256, 2)
void gemm_bf16_bias(const __bf16* __restrict__ A, const __bf16* __restrict__ W,
                    const float* __restrict__ bias, float* __restrict__ Y,
                    int M, int N, int K) {
    int tid  = threadIdx.x;
    int lane = tid & 31;
    int wid  = tid >> 5;
    int wm   = wid & 3;      // 0..3  (row quadrant, 32 rows each)
    int wn   = wid >> 2;     // 0..1  (col half, 64 cols each)

    int m0 = blockIdx.y * TILE_M;
    int n0 = blockIdx.x * TILE_N;

    v8f acc[2][4] = {};

#if USE_TDM
    __shared__ __bf16 sA[2][TILE_M * LDK];
    __shared__ __bf16 sB[2][TILE_N * LDK];

    int ntiles = K / TILE_K;
    if (wid == 0) {
        tdm_load_tile(lds_byte_off(sA[0]), A + (size_t)m0 * K, TILE_M, K, M);
        tdm_load_tile(lds_byte_off(sB[0]), W + (size_t)n0 * K, TILE_N, K, N);
    }
    for (int it = 0; it < ntiles; ++it) {
        if (wid == 0) __builtin_amdgcn_s_wait_tensorcnt(0);  // tile `it` landed
        __syncthreads();                                     // visible to all
        int buf = it & 1;
        if (wid == 0 && it + 1 < ntiles) {
            // DMA next tile into the idle buffer while we compute this one.
            int kk = (it + 1) * TILE_K;
            tdm_load_tile(lds_byte_off(sA[buf ^ 1]),
                          A + (size_t)m0 * K + kk, TILE_M, K, M);
            tdm_load_tile(lds_byte_off(sB[buf ^ 1]),
                          W + (size_t)n0 * K + kk, TILE_N, K, N);
        }
        const __bf16* cA = sA[buf];
        const __bf16* cB = sB[buf];

        v16bf afr[2], bfr[4];
#pragma unroll
        for (int i = 0; i < 2; ++i)
            afr[i] = load_a_frag(cA + (wm * 32 + i * 16) * LDK, lane);
#pragma unroll
        for (int j = 0; j < 4; ++j)
            bfr[j] = load_b_frag(cB + (wn * 64 + j * 16) * LDK, lane);

#pragma unroll
        for (int i = 0; i < 2; ++i)
#pragma unroll
            for (int j = 0; j < 4; ++j)
                acc[i][j] = __builtin_amdgcn_wmma_f32_16x16x32_bf16(
                    false, afr[i], false, bfr[j], (short)0, acc[i][j],
                    false, false);
    }
#else
    __shared__ __bf16 sA[TILE_M * LDK];
    __shared__ __bf16 sB[TILE_N * LDK];

    int r    = tid >> 1;     // 0..127: tile row handled by this thread
    int half = tid & 1;      // which 16-element chunk of the 32-wide K slice

    for (int kk = 0; kk < K; kk += TILE_K) {
        const __bf16* ga = A + (size_t)(m0 + r) * K + kk + half * 16;
        const __bf16* gb = W + (size_t)(n0 + r) * K + kk + half * 16;
        uint4 a0 = *(const uint4*)(ga);
        uint4 a1 = *(const uint4*)(ga + 8);
        uint4 b0 = *(const uint4*)(gb);
        uint4 b1 = *(const uint4*)(gb + 8);
        if (kk + TILE_K < K) {
            __builtin_prefetch(ga + TILE_K, 0, 1);
            __builtin_prefetch(gb + TILE_K, 0, 1);
        }
        __bf16* la = sA + r * LDK + half * 16;
        __bf16* lb = sB + r * LDK + half * 16;
        *(uint4*)(la)     = a0;
        *(uint4*)(la + 8) = a1;
        *(uint4*)(lb)     = b0;
        *(uint4*)(lb + 8) = b1;
        __syncthreads();

        v16bf afr[2], bfr[4];
#pragma unroll
        for (int i = 0; i < 2; ++i)
            afr[i] = load_a_frag(sA + (wm * 32 + i * 16) * LDK, lane);
#pragma unroll
        for (int j = 0; j < 4; ++j)
            bfr[j] = load_b_frag(sB + (wn * 64 + j * 16) * LDK, lane);

#pragma unroll
        for (int i = 0; i < 2; ++i)
#pragma unroll
            for (int j = 0; j < 4; ++j)
                acc[i][j] = __builtin_amdgcn_wmma_f32_16x16x32_bf16(
                    false, afr[i], false, bfr[j], (short)0, acc[i][j],
                    false, false);

        __syncthreads();
    }
#endif

    // ---- store: D layout, lane L -> col n = L&15, VGPR r -> row r + 8*(L>>4)
    int cn = lane & 15;
    int mh = (lane >> 4) * 8;
#pragma unroll
    for (int i = 0; i < 2; ++i) {
#pragma unroll
        for (int j = 0; j < 4; ++j) {
            int col = n0 + wn * 64 + j * 16 + cn;
            float bb = bias ? bias[col] : 0.0f;
#pragma unroll
            for (int rr = 0; rr < 8; ++rr) {
                int row = m0 + wm * 32 + i * 16 + mh + rr;
                Y[(size_t)row * N + col] = acc[i][j][rr] + bb;
            }
        }
    }
}

// ---------------------------------------------------------------------------
// float -> bf16 convert (grid-stride)
// ---------------------------------------------------------------------------
__global__ void cvt_bf16_kernel(const float* __restrict__ in,
                                __bf16* __restrict__ out, int n) {
    int i = blockIdx.x * blockDim.x + threadIdx.x;
    int stride = gridDim.x * blockDim.x;
    for (; i < n; i += stride) out[i] = (__bf16)in[i];
}

// ---------------------------------------------------------------------------
// key_scores[t] = x[t,:] . Wks + bks   (one wave per row)
// ---------------------------------------------------------------------------
__global__ void key_scores_kernel(const float* __restrict__ x,
                                  const float* __restrict__ Wks,
                                  const float* __restrict__ bks,
                                  float* __restrict__ scores, int T, int C) {
    int w    = (blockIdx.x * blockDim.x + threadIdx.x) >> 5;
    int lane = threadIdx.x & 31;
    if (w >= T) return;
    const float* xr = x + (size_t)w * C;
    float acc = 0.f;
    for (int c = lane; c < C; c += 32) acc += xr[c] * Wks[c];
    for (int m = 16; m; m >>= 1) acc += __shfl_xor(acc, m, 32);
    if (lane == 0) scores[w] = acc + bks[0];
}

// ---------------------------------------------------------------------------
// Top-64 of scores[0..t] per query t (values desc, ties -> lowest index).
// When fewer than 64 candidates, tail indices clamp to t (JAX semantics).
// One wave per t; scores vector (8KB) stays cache-resident.
// ---------------------------------------------------------------------------
__global__ void topk_kernel(const float* __restrict__ scores,
                            int* __restrict__ topk, int T, int KSEL) {
    int w    = (blockIdx.x * blockDim.x + threadIdx.x) >> 5;
    int lane = threadIdx.x & 31;
    if (w >= T) return;
    int t    = w;
    int cand = t + 1;
    int kv   = KSEL < cand ? KSEL : cand;

    float prev_v = __int_as_float(0x7f800000);  // +inf
    int   prev_i = -1;

    for (int s = 0; s < KSEL; ++s) {
        float best_v = __int_as_float(0xff800000);  // -inf
        int   best_i = 0x7fffffff;
        for (int i = lane; i <= t; i += 32) {
            float v = scores[i];
            bool ok = (v < prev_v) || (v == prev_v && i > prev_i);
            if (ok && (v > best_v || (v == best_v && i < best_i))) {
                best_v = v;
                best_i = i;
            }
        }
        for (int m = 16; m; m >>= 1) {
            float ov = __shfl_xor(best_v, m, 32);
            int   oi = __shfl_xor(best_i, m, 32);
            if (ov > best_v || (ov == best_v && oi < best_i)) {
                best_v = ov;
                best_i = oi;
            }
        }
        int sel = (s < kv) ? best_i : t;
        if (s < kv) { prev_v = best_v; prev_i = best_i; }
        if (lane == 0) topk[(size_t)t * KSEL + s] = sel;
    }
}

// ---------------------------------------------------------------------------
// Gathered attention: one wave per (head h, query t). hs = 64 = 2 dims/lane.
// Phase 1: logits via lane-pair dot + wave reduction; Phase 2: softmax-weighted
// gather of V. Writes att (bf16) for the final WMMA projection.
// ---------------------------------------------------------------------------
__global__ void attn_kernel(const float* __restrict__ q,
                            const float* __restrict__ k,
                            const float* __restrict__ v,
                            const int* __restrict__ topk,
                            __bf16* __restrict__ att,
                            int T, int C, int nh, int KSEL) {
    int hs   = C / nh;   // 64
    int lane = threadIdx.x & 31;
    int w    = (blockIdx.x * blockDim.x + threadIdx.x) >> 5;
    if (w >= T * nh) return;
    int h = w % nh;
    int t = w / nh;
    int d0 = lane, d1 = lane + 32;

    const float* qr = q + (size_t)t * C + h * hs;
    float q0 = qr[d0], q1 = qr[d1];
    const int* ti = topk + (size_t)t * KSEL;
    float scale = rsqrtf((float)hs);

    float l0 = 0.f, l1 = 0.f;
    for (int s = 0; s < KSEL; ++s) {
        int idx = ti[s];
        const float* kr = k + (size_t)idx * C + h * hs;
        float partial = q0 * kr[d0] + q1 * kr[d1];
        for (int m = 16; m; m >>= 1) partial += __shfl_xor(partial, m, 32);
        float logit = partial * scale;
        if (lane == (s & 31)) {
            if (s < 32) l0 = logit; else l1 = logit;
        }
    }

    float mx = fmaxf(l0, l1);
    for (int m = 16; m; m >>= 1) mx = fmaxf(mx, __shfl_xor(mx, m, 32));
    float e0 = __expf(l0 - mx), e1 = __expf(l1 - mx);
    float den = e0 + e1;
    for (int m = 16; m; m >>= 1) den += __shfl_xor(den, m, 32);
    float inv = 1.0f / den;

    float a0 = 0.f, a1 = 0.f;
    for (int s = 0; s < KSEL; ++s) {
        int idx = ti[s];
        float es = __shfl((s < 32) ? e0 : e1, s & 31, 32);
        float p = es * inv;
        const float* vr = v + (size_t)idx * C + h * hs;
        a0 += p * vr[d0];
        a1 += p * vr[d1];
    }

    __bf16* ar = att + (size_t)t * C + h * hs;
    ar[d0] = (__bf16)a0;
    ar[d1] = (__bf16)a1;
}

// ---------------------------------------------------------------------------
// Host-side orchestration
// ---------------------------------------------------------------------------
extern "C" void kernel_launch(void* const* d_in, const int* in_sizes, int n_in,
                              void* d_out, int out_size, void* d_ws, size_t ws_size,
                              hipStream_t stream) {
    const float* x   = (const float*)d_in[0];
    const float* Wq  = (const float*)d_in[1];
    const float* bq  = (const float*)d_in[2];
    const float* Wk  = (const float*)d_in[3];
    const float* bk  = (const float*)d_in[4];
    const float* Wv  = (const float*)d_in[5];
    const float* bv  = (const float*)d_in[6];
    const float* Wo  = (const float*)d_in[7];
    const float* bo  = (const float*)d_in[8];
    const float* Wks = (const float*)d_in[9];
    const float* bks = (const float*)d_in[10];

    int C = 1;
    while ((long long)C * C < (long long)in_sizes[1]) C <<= 1;  // 1024
    int T  = in_sizes[0] / C;                                   // 2048
    const int NH   = 16;
    const int KSEL = 64;

    size_t TC = (size_t)T * C;
    size_t CC = (size_t)C * C;

    char* p = (char*)d_ws;
    __bf16* xb  = (__bf16*)p; p += TC * sizeof(__bf16);
    __bf16* wqb = (__bf16*)p; p += CC * sizeof(__bf16);
    __bf16* wkb = (__bf16*)p; p += CC * sizeof(__bf16);
    __bf16* wvb = (__bf16*)p; p += CC * sizeof(__bf16);
    __bf16* wob = (__bf16*)p; p += CC * sizeof(__bf16);
    float*  qf  = (float*)p;  p += TC * sizeof(float);
    float*  kf  = (float*)p;  p += TC * sizeof(float);
    float*  vf  = (float*)p;  p += TC * sizeof(float);
    float*  sc  = (float*)p;  p += (size_t)T * sizeof(float);
    int*    tk  = (int*)p;    p += (size_t)T * KSEL * sizeof(int);
    __bf16* att = (__bf16*)p;

    // 1) f32 -> bf16 conversions
    {
        int n = (int)TC;
        int grid = (n + 1023) / 1024;
        cvt_bf16_kernel<<<grid, 256, 0, stream>>>(x, xb, n);
        int nw = (int)CC;
        int gw = (nw + 1023) / 1024;
        cvt_bf16_kernel<<<gw, 256, 0, stream>>>(Wq, wqb, nw);
        cvt_bf16_kernel<<<gw, 256, 0, stream>>>(Wk, wkb, nw);
        cvt_bf16_kernel<<<gw, 256, 0, stream>>>(Wv, wvb, nw);
        cvt_bf16_kernel<<<gw, 256, 0, stream>>>(Wo, wob, nw);
    }

    // 2) key scores
    key_scores_kernel<<<(T + 3) / 4, 128, 0, stream>>>(x, Wks, bks, sc, T, C);

    // 3) top-k indices
    topk_kernel<<<(T + 3) / 4, 128, 0, stream>>>(sc, tk, T, KSEL);

    // 4) Q/K/V projections (WMMA GEMM, bias fused, TDM-fed tiles)
    dim3 ggrid(C / TILE_N, T / TILE_M);
    gemm_bf16_bias<<<ggrid, 256, 0, stream>>>(xb, wqb, bq, qf, T, C, C);
    gemm_bf16_bias<<<ggrid, 256, 0, stream>>>(xb, wkb, bk, kf, T, C, C);
    gemm_bf16_bias<<<ggrid, 256, 0, stream>>>(xb, wvb, bv, vf, T, C, C);

    // 5) sparse gathered attention -> att (bf16)
    {
        int waves  = T * NH;
        int blocks = (waves + 7) / 8;   // 8 waves per 256-thread block
        attn_kernel<<<blocks, 256, 0, stream>>>(qf, kf, vf, tk, att, T, C, NH, KSEL);
    }

    // 6) output projection -> d_out (f32)
    gemm_bf16_bias<<<ggrid, 256, 0, stream>>>(att, wob, bo, (float*)d_out, T, C, C);
}
